// kNNHyperbolicAttentionLayer_8924942041790
// MI455X (gfx1250) — compile-verified, matrix operand order checked by reading, not run on hardware
//
#include <hip/hip_runtime.h>
#include <hip/hip_bf16.h>
#include <math.h>

typedef __attribute__((ext_vector_type(16))) _Float16 v16h;
typedef __attribute__((ext_vector_type(8)))  float    v8f;

#define DIMC   256
#define HEADSC 8
#define HDC    32
#define KNNC   16
#define BSZC   4
#define NSEQC  4096
#define ROWSC  (BSZC * NSEQC)          // 16384
#define FFC    (4 * DIMC)              // 1024
#define BKC    256                     // K-chunk staged in LDS

// ---------------------------------------------------------------------------
// CDNA5 async global->LDS copy (ASYNCcnt-tracked), via inline asm (portable
// across toolchains; builtin signatures differ between clang-22/23).
// VDST = per-lane LDS byte address, VADDR = 64-bit global address (GV mode).
// ---------------------------------------------------------------------------
__device__ __forceinline__ void async_g2l_b128(void* lds_dst, const void* gsrc) {
  unsigned lds_off = (unsigned)(unsigned long long)(size_t)lds_dst;
  asm volatile("global_load_async_to_lds_b128 %0, %1, off"
               :
               : "v"(lds_off), "v"((unsigned long long)(size_t)gsrc)
               : "memory");
}

__device__ __forceinline__ void wait_asynccnt0() {
  asm volatile("s_wait_asynccnt 0x0" ::: "memory");
}

// ---------------------------------------------------------------------------
// Weight convert + transpose:  W[K,N] f32 -> Wt[N,K] f16  (contiguous K rows
// so WMMA B fragments are 32-byte contiguous per lane)
// ---------------------------------------------------------------------------
__global__ __launch_bounds__(256) void wcvt_kernel(const float* __restrict__ W,
                                                   _Float16* __restrict__ Wt,
                                                   int K, int N) {
  long t = (long)blockIdx.x * 256 + threadIdx.x;
  if (t >= (long)N * K) return;
  int n  = (int)(t / K);
  int kk = (int)(t % K);
  Wt[t] = (_Float16)W[(long)kk * N + n];
}

// ---------------------------------------------------------------------------
// Fused LayerNorm (+ optional time embedding) -> f16 rows.
// One wave32 per 256-wide row, 8 rows per 256-thread block.
// ---------------------------------------------------------------------------
__global__ __launch_bounds__(256) void ln_kernel(const float* __restrict__ x,
                                                 const float* __restrict__ w,
                                                 const float* __restrict__ b,
                                                 const float* __restrict__ temb,
                                                 _Float16* __restrict__ out,
                                                 int Nn) {
  int wave = threadIdx.x >> 5, lane = threadIdx.x & 31;
  long row = (long)blockIdx.x * 8 + wave;          // over B*N
  const float* xr = x + row * DIMC;
  float v[8];
  float s = 0.f;
#pragma unroll
  for (int u = 0; u < 8; ++u) { v[u] = xr[u * 32 + lane]; s += v[u]; }
#pragma unroll
  for (int m = 16; m >= 1; m >>= 1) s += __shfl_xor(s, m, 32);
  float mean = s * (1.0f / DIMC);
  float vs = 0.f;
#pragma unroll
  for (int u = 0; u < 8; ++u) { float d = v[u] - mean; vs += d * d; }
#pragma unroll
  for (int m = 16; m >= 1; m >>= 1) vs += __shfl_xor(vs, m, 32);
  float inv = rsqrtf(vs * (1.0f / DIMC) + 1e-5f);
  const float* te = temb ? (temb + (row / Nn) * DIMC) : nullptr;
#pragma unroll
  for (int u = 0; u < 8; ++u) {
    int c = u * 32 + lane;
    float hv = (v[u] - mean) * inv * w[c] + b[c];
    if (te) hv += te[c];
    out[row * DIMC + c] = (_Float16)hv;
  }
}

// ---------------------------------------------------------------------------
// WMMA GEMM: C[M,N] = A[M,K](f16) * Bt[N,K](f16)^T + bias, templated epilogue.
//
// Block = 256 threads = 8 waves arranged 4(M) x 2(N); block tile 128x64.
// Wave tile 32x32 = four 16x16 WMMA tiles (2 A frags x 2 B frags per K-step).
// The 64-col B panel is staged per 256-wide K-chunk into LDS with
// GLOBAL_LOAD_ASYNC_TO_LDS_B128 (ASYNCcnt) and read back via ds_load_b128;
// A fragments stream from global (L2-resident activations).
// ---------------------------------------------------------------------------
#define EPI_STORE_F32     0
#define EPI_RES_STORE_F32 1
#define EPI_GELU_F16      2
#define EPI_ADD_INPLACE   3

__device__ __forceinline__ float gelu_exact(float x) {
  return 0.5f * x * (1.0f + erff(x * 0.70710678118f));
}

template <int EPI>
__global__ __launch_bounds__(256) void wmma_gemm(const _Float16* __restrict__ A,
                                                 const _Float16* __restrict__ Bt,
                                                 const float* __restrict__ bias,
                                                 const float* __restrict__ resid,
                                                 float* __restrict__ Cf,
                                                 _Float16* __restrict__ Ch,
                                                 int Md, int Nd, int Kd) {
  __shared__ __align__(32) _Float16 Bs[64 * BKC];   // 32 KB B panel chunk

  int tid  = threadIdx.x;
  int wave = tid >> 5, lane = tid & 31;
  int wm = (wave >> 1) * 32;                 // 0,32,64,96
  int wn = (wave & 1) * 32;                  // 0,32
  long tileM  = (long)blockIdx.y * 128 + wm;
  long tileNb = (long)blockIdx.x * 64;       // block N base
  int lhalf = lane >> 4;                     // K-half select (contiguous k = 16*lhalf + s)
  int lmod  = lane & 15;

  const _Float16* Ar0 = A + (tileM + lmod)      * (long)Kd + lhalf * 16;
  const _Float16* Ar1 = A + (tileM + 16 + lmod) * (long)Kd + lhalf * 16;
  const _Float16* Bl0 = Bs + (wn + lmod)      * BKC + lhalf * 16;
  const _Float16* Bl1 = Bs + (wn + 16 + lmod) * BKC + lhalf * 16;

  v8f acc00 = {}, acc01 = {}, acc10 = {}, acc11 = {};

  for (int kk0 = 0; kk0 < Kd; kk0 += BKC) {
    // ---- stage 64 x BKC B panel chunk into LDS (16B per lane per issue) ----
    {
      const int chunks = 64 * (BKC * 2 / 16);          // 2048
#pragma unroll
      for (int c0 = 0; c0 < chunks; c0 += 256) {
        int cid = c0 + tid;
        int r   = cid >> 5;                            // row in panel
        int ib  = (cid & 31) * 16;                     // byte within row
        const char* src = (const char*)(Bt + (tileNb + r) * (long)Kd + kk0) + ib;
        char*       dst = (char*)Bs + r * (BKC * 2) + ib;
        async_g2l_b128(dst, src);
      }
      wait_asynccnt0();
      __syncthreads();
    }

    // ---- compute over the chunk ----
#pragma unroll 2
    for (int k = 0; k < BKC; k += 32) {
      v16h a0 = *reinterpret_cast<const v16h*>(Ar0 + kk0 + k);
      v16h a1 = *reinterpret_cast<const v16h*>(Ar1 + kk0 + k);
      v16h b0 = *reinterpret_cast<const v16h*>(Bl0 + k);
      v16h b1 = *reinterpret_cast<const v16h*>(Bl1 + k);
      if (k + 64 < BKC) {
        __builtin_prefetch(Ar0 + kk0 + k + 64, 0, 3);
        __builtin_prefetch(Ar1 + kk0 + k + 64, 0, 3);
      }
      acc00 = __builtin_amdgcn_wmma_f32_16x16x32_f16(false, a0, false, b0,
                                                     (short)0, acc00, false, false);
      acc01 = __builtin_amdgcn_wmma_f32_16x16x32_f16(false, a0, false, b1,
                                                     (short)0, acc01, false, false);
      acc10 = __builtin_amdgcn_wmma_f32_16x16x32_f16(false, a1, false, b0,
                                                     (short)0, acc10, false, false);
      acc11 = __builtin_amdgcn_wmma_f32_16x16x32_f16(false, a1, false, b1,
                                                     (short)0, acc11, false, false);
    }
    __syncthreads();   // protect LDS panel before next-stage overwrite
  }

  // C/D layout: VGPR r -> M = r + 8*(lane>=16); N = lane&15
#pragma unroll
  for (int r = 0; r < 8; ++r) {
    long row0 = tileM + lhalf * 8 + r;
    long row1 = row0 + 16;
    long col0 = tileNb + wn + lmod;
    long col1 = col0 + 16;
    float b0v = bias[col0];
    float b1v = bias[col1];
    float c00 = acc00[r] + b0v, c01 = acc01[r] + b1v;
    float c10 = acc10[r] + b0v, c11 = acc11[r] + b1v;
    long i00 = row0 * Nd + col0, i01 = row0 * Nd + col1;
    long i10 = row1 * Nd + col0, i11 = row1 * Nd + col1;
    if constexpr (EPI == EPI_STORE_F32) {
      Cf[i00] = c00; Cf[i01] = c01; Cf[i10] = c10; Cf[i11] = c11;
    } else if constexpr (EPI == EPI_RES_STORE_F32) {
      Cf[i00] = resid[i00] + c00; Cf[i01] = resid[i01] + c01;
      Cf[i10] = resid[i10] + c10; Cf[i11] = resid[i11] + c11;
    } else if constexpr (EPI == EPI_GELU_F16) {
      Ch[i00] = (_Float16)gelu_exact(c00); Ch[i01] = (_Float16)gelu_exact(c01);
      Ch[i10] = (_Float16)gelu_exact(c10); Ch[i11] = (_Float16)gelu_exact(c11);
    } else {
      Cf[i00] += c00; Cf[i01] += c01; Cf[i10] += c10; Cf[i11] += c11;
    }
  }
}

// ---------------------------------------------------------------------------
// Poincare kNN: one wave per query row. Each lane scans N/32 candidates with a
// register-resident branch-free sorted top-16, then a 16-round LDS + shfl_xor
// argmin merge produces the global ascending top-16 (dist + index).
// ---------------------------------------------------------------------------
__global__ __launch_bounds__(128) void knn_kernel(const float* __restrict__ pos,
                                                  const float* __restrict__ cptr,
                                                  float* __restrict__ dstk,
                                                  int* __restrict__ idxk,
                                                  int Nn) {
  __shared__ float sd[4][32][KNNC];
  __shared__ int   si[4][32][KNNC];
  int wave = threadIdx.x >> 5, lane = threadIdx.x & 31;
  long qi = (long)blockIdx.x * 4 + wave;   // over B*N (exact grid)
  int b = (int)(qi / Nn), i = (int)(qi % Nn);
  float c = cptr[0];
  float rsc = rsqrtf(c);
  const float* pb = pos + (long)b * Nn * 3;
  float xi = pb[(long)i * 3 + 0], yi = pb[(long)i * 3 + 1], zi = pb[(long)i * 3 + 2];
  float sqi = xi * xi + yi * yi + zi * zi;
  float deni = 1.0f - c * sqi;

  float di[KNNC];
  int   ii[KNNC];
#pragma unroll
  for (int s = 0; s < KNNC; ++s) { di[s] = 3.4e38f; ii[s] = 0; }

  for (int j = lane; j < Nn; j += 32) {
    float xj = pb[(long)j * 3 + 0], yj = pb[(long)j * 3 + 1], zj = pb[(long)j * 3 + 2];
    float sqj = xj * xj + yj * yj + zj * zj;
    float dot = xi * xj + yi * yj + zi * zj;
    float diff2 = fmaxf(sqi + sqj - 2.0f * dot, 0.0f);
    float num = 2.0f * c * diff2;
    float den = deni * (1.0f - c * sqj);
    float arg = fmaxf(1.0f + num / (den + 1e-8f), 1.0f);
    float d = logf(arg + sqrtf(fmaxf(arg * arg - 1.0f, 0.0f))) * rsc;
    if (d < di[KNNC - 1]) {
#pragma unroll
      for (int s = KNNC - 1; s >= 1; --s) {
        if (d < di[s - 1])      { di[s] = di[s - 1]; ii[s] = ii[s - 1]; }
        else if (d < di[s])     { di[s] = d;         ii[s] = j; }
      }
      if (d < di[0]) { di[0] = d; ii[0] = j; }
    }
  }

#pragma unroll
  for (int s = 0; s < KNNC; ++s) { sd[wave][lane][s] = di[s]; si[wave][lane][s] = ii[s]; }
  __syncthreads();

  int p = 0;
  for (int t = 0; t < KNNC; ++t) {
    float cand = (p < KNNC) ? sd[wave][lane][p] : 3.4e38f;
    int   cid  = (p < KNNC) ? si[wave][lane][p] : -1;
    int   cl   = lane;
#pragma unroll
    for (int m = 16; m >= 1; m >>= 1) {
      float od = __shfl_xor(cand, m, 32);
      int   oi = __shfl_xor(cid,  m, 32);
      int   ol = __shfl_xor(cl,   m, 32);
      if (od < cand) { cand = od; cid = oi; cl = ol; }
    }
    if (lane == cl) p++;
    if (lane == 0) { dstk[qi * KNNC + t] = cand; idxk[qi * KNNC + t] = cid; }
  }
}

// ---------------------------------------------------------------------------
// Sparse kNN attention: one wave per (b,i,h). Phase 1: lanes 0..15 each own a
// neighbor (dot over HD=32 + softmax via shfl). Phase 2: lanes own output dims
// and accumulate prob-weighted V gathers (coalesced 32-float rows).
// ---------------------------------------------------------------------------
__global__ __launch_bounds__(256) void attn_kernel(const float* __restrict__ q,
                                                   const float* __restrict__ kf,
                                                   const float* __restrict__ vf,
                                                   const float* __restrict__ dstk,
                                                   const int* __restrict__ idxk,
                                                   const float* __restrict__ log_tau,
                                                   _Float16* __restrict__ out16,
                                                   int Nn) {
  __shared__ float sp[8][KNNC];
  __shared__ int   sj[8][KNNC];
  int wave = threadIdx.x >> 5, lane = threadIdx.x & 31;
  long w  = (long)blockIdx.x * 8 + wave;     // over B*N*H (exact grid)
  int  h  = (int)(w % HEADSC);
  long bi = w / HEADSC;                      // row in [B*N)
  long bbase = (bi / Nn) * (long)Nn;         // batch row base

  float tau = __expf(log_tau[0]) + 1e-8f;
  const float* qrow = q + bi * DIMC + h * HDC;

  float score = -3.4e38f;
  int j = 0;
  if (lane < KNNC) {
    j = idxk[bi * KNNC + lane];
    const float* krow = kf + (bbase + j) * DIMC + h * HDC;
    float dot = 0.f;
#pragma unroll
    for (int d = 0; d < HDC; ++d) dot += qrow[d] * krow[d];
    score = dot * 0.17677669529f - dstk[bi * KNNC + lane] / tau;
  }
  // softmax over the 16-lane half (masks 8..1 stay within each half)
  float m = score;
#pragma unroll
  for (int s = 8; s >= 1; s >>= 1) m = fmaxf(m, __shfl_xor(m, s, 32));
  float e = (lane < KNNC) ? __expf(score - m) : 0.f;
  float ssum = e;
#pragma unroll
  for (int s = 8; s >= 1; s >>= 1) ssum += __shfl_xor(ssum, s, 32);
  if (lane < KNNC) { sp[wave][lane] = e / ssum; sj[wave][lane] = j; }
  __syncthreads();

  float accd = 0.f;
  const float* vb = vf + bbase * DIMC + h * HDC + lane;
#pragma unroll
  for (int kk = 0; kk < KNNC; ++kk)
    accd += sp[wave][kk] * vb[(long)sj[wave][kk] * DIMC];
  out16[bi * DIMC + h * HDC + lane] = (_Float16)accd;
}

// ---------------------------------------------------------------------------
// Host-side orchestration
// ---------------------------------------------------------------------------
extern "C" void kernel_launch(void* const* d_in, const int* in_sizes, int n_in,
                              void* d_out, int out_size, void* d_ws, size_t ws_size,
                              hipStream_t stream) {
  (void)in_sizes; (void)n_in; (void)out_size; (void)ws_size;
  const float* x        = (const float*)d_in[0];
  const float* pos      = (const float*)d_in[1];
  const float* cptr     = (const float*)d_in[2];
  const float* temb     = (const float*)d_in[3];
  const float* Wq       = (const float*)d_in[4];
  const float* bq       = (const float*)d_in[5];
  const float* Wk       = (const float*)d_in[6];
  const float* bk       = (const float*)d_in[7];
  const float* Wv       = (const float*)d_in[8];
  const float* bv       = (const float*)d_in[9];
  const float* Wo       = (const float*)d_in[10];
  const float* bo       = (const float*)d_in[11];
  const float* W1       = (const float*)d_in[12];
  const float* b1       = (const float*)d_in[13];
  const float* W2       = (const float*)d_in[14];
  const float* b2       = (const float*)d_in[15];
  const float* ln1_w    = (const float*)d_in[16];
  const float* ln1_b    = (const float*)d_in[17];
  const float* ln2_w    = (const float*)d_in[18];
  const float* ln2_b    = (const float*)d_in[19];
  const float* log_tau  = (const float*)d_in[20];
  float* out = (float*)d_out;

  char* wsb = (char*)d_ws;
  size_t off = 0;
  auto alloc = [&](size_t bytes) -> void* {
    off = (off + 255) & ~(size_t)255;
    void* p = wsb + off;
    off += bytes;
    return p;
  };
  _Float16* h16   = (_Float16*)alloc((size_t)ROWSC * DIMC * 2);
  float*    qf    = (float*)alloc((size_t)ROWSC * DIMC * 4);
  float*    kfp   = (float*)alloc((size_t)ROWSC * DIMC * 4);
  float*    vfp   = (float*)alloc((size_t)ROWSC * DIMC * 4);
  _Float16* at16  = (_Float16*)alloc((size_t)ROWSC * DIMC * 2);
  _Float16* h216  = (_Float16*)alloc((size_t)ROWSC * DIMC * 2);
  _Float16* ff16  = (_Float16*)alloc((size_t)ROWSC * FFC * 2);
  float*    dstk  = (float*)alloc((size_t)ROWSC * KNNC * 4);
  int*      idxk  = (int*)alloc((size_t)ROWSC * KNNC * 4);
  _Float16* Wq16  = (_Float16*)alloc((size_t)DIMC * DIMC * 2);
  _Float16* Wk16  = (_Float16*)alloc((size_t)DIMC * DIMC * 2);
  _Float16* Wv16  = (_Float16*)alloc((size_t)DIMC * DIMC * 2);
  _Float16* Wo16  = (_Float16*)alloc((size_t)DIMC * DIMC * 2);
  _Float16* W116  = (_Float16*)alloc((size_t)DIMC * FFC * 2);
  _Float16* W216  = (_Float16*)alloc((size_t)FFC * DIMC * 2);

  // --- weight convert + transpose (f32 -> f16, [K,N] -> [N,K]) ---
  {
    int n1 = DIMC * DIMC;
    wcvt_kernel<<<(n1 + 255) / 256, 256, 0, stream>>>(Wq, Wq16, DIMC, DIMC);
    wcvt_kernel<<<(n1 + 255) / 256, 256, 0, stream>>>(Wk, Wk16, DIMC, DIMC);
    wcvt_kernel<<<(n1 + 255) / 256, 256, 0, stream>>>(Wv, Wv16, DIMC, DIMC);
    wcvt_kernel<<<(n1 + 255) / 256, 256, 0, stream>>>(Wo, Wo16, DIMC, DIMC);
    int n2 = DIMC * FFC;
    wcvt_kernel<<<(n2 + 255) / 256, 256, 0, stream>>>(W1, W116, DIMC, FFC);
    wcvt_kernel<<<(n2 + 255) / 256, 256, 0, stream>>>(W2, W216, FFC, DIMC);
  }

  // --- kNN over Poincare distances (independent of GEMM pipeline) ---
  knn_kernel<<<ROWSC / 4, 128, 0, stream>>>(pos, cptr, dstk, idxk, NSEQC);

  // --- h = LN1(x) + time_emb  (f16) ---
  ln_kernel<<<ROWSC / 8, 256, 0, stream>>>(x, ln1_w, ln1_b, temb, h16, NSEQC);

  // --- Q,K,V projections via WMMA (block tile 128x64) ---
  dim3 g256(DIMC / 64, ROWSC / 128);
  wmma_gemm<EPI_STORE_F32><<<g256, 256, 0, stream>>>(h16, Wq16, bq, nullptr, qf, nullptr, ROWSC, DIMC, DIMC);
  wmma_gemm<EPI_STORE_F32><<<g256, 256, 0, stream>>>(h16, Wk16, bk, nullptr, kfp, nullptr, ROWSC, DIMC, DIMC);
  wmma_gemm<EPI_STORE_F32><<<g256, 256, 0, stream>>>(h16, Wv16, bv, nullptr, vfp, nullptr, ROWSC, DIMC, DIMC);

  // --- sparse 16-neighbor attention ---
  attn_kernel<<<(ROWSC * HEADSC) / 8, 256, 0, stream>>>(qf, kfp, vfp, dstk, idxk, log_tau, at16, NSEQC);

  // --- x2 = x + attn @ Wo + bo  -> d_out ---
  wmma_gemm<EPI_RES_STORE_F32><<<g256, 256, 0, stream>>>(at16, Wo16, bo, x, out, nullptr, ROWSC, DIMC, DIMC);

  // --- h2 = LN2(x2)  (f16) ---
  ln_kernel<<<ROWSC / 8, 256, 0, stream>>>(out, ln2_w, ln2_b, nullptr, h216, NSEQC);

  // --- ff1 = gelu(h2 @ W1 + b1)  (f16) ---
  dim3 gff(FFC / 64, ROWSC / 128);
  wmma_gemm<EPI_GELU_F16><<<gff, 256, 0, stream>>>(h216, W116, b1, nullptr, nullptr, ff16, ROWSC, FFC, DIMC);

  // --- out = x2 + ff1 @ W2 + b2  (accumulate into d_out) ---
  wmma_gemm<EPI_ADD_INPLACE><<<g256, 256, 0, stream>>>(ff16, W216, b2, nullptr, out, nullptr, ROWSC, DIMC, FFC);
}
